// _VectorQuantizer_64201171140812
// MI455X (gfx1250) — matrix operand order repaired and checked by reading, not compiled, not run on hardware
//
#include <hip/hip_runtime.h>

// ---------------------------------------------------------------------------
// Grouped vector quantizer (wav2vec2-style), fused:
//   logits = x @ W_cb + b   (bf16 WMMA, f32 acc)  -> LDS
//   softmax stats + argmax  (f32, shfl reductions)
//   cv     = P @ codevectors (bf16 WMMA, f32 acc), row-scaled by 1/sum, masked
// One wave = one token (16 groups = 16-row WMMA M-tile).
// ---------------------------------------------------------------------------

typedef __bf16 bf16;
typedef __attribute__((ext_vector_type(16))) __bf16 v16bf;
typedef __attribute__((ext_vector_type(8)))  float  v8f;

#define B_   8
#define S_   2048
#define H_   1024
#define G_   16
#define K_   320
#define D_   64
#define NTOK (B_ * S_)        // 16384 tokens
#define NROW (NTOK * G_)      // 262144 rows
#define LP   328              // padded LDS logits row stride (floats), 16B-aligned runs
#define WPB  2                // waves per block

__device__ __forceinline__ bf16 f2bf(float f) {
  union { float f; unsigned u; } c; c.f = f;
  unsigned r = c.u + 0x7FFFu + ((c.u >> 16) & 1u);   // round-to-nearest-even
  unsigned short h = (unsigned short)(r >> 16);
  bf16 b;
  __builtin_memcpy(&b, &h, sizeof(h));
  return b;
}

// ---------------------------------------------------------------------------
// Kernel 1: pre-swizzle W_cb [64x320] and codevectors [320x64] (f32) into
// bf16 B-operand tiles for v_wmma_f32_16x16x32_bf16.
// B-tile layout per wave: lane L (n=L&15, hi=L>>4), element e holds
// Bmat[K = 8*hi + e + (e >= 8 ? 8 : 0)][n]. Stored flat as [tile][lane][e] so the hot loop
// does one contiguous 32-byte load per lane.
//   Wsw: tiles (t=0..19, kk=0..1): Bmat = W_cb[32*kk + krow][16*t + n]
//   Vsw: tiles (kt=0..9, nt=0..3): Bmat = codevectors[32*kt + krow][16*nt + n]
// ---------------------------------------------------------------------------
__global__ void vq_swizzle(const float* __restrict__ W, const float* __restrict__ V,
                           bf16* __restrict__ Wsw, bf16* __restrict__ Vsw) {
  int i = blockIdx.x * blockDim.x + threadIdx.x;
  if (i < 20480) {                       // 20*2 tiles * 32 lanes * 16 elems
    int e = i & 15, lane = (i >> 4) & 31, tk = i >> 9;
    int kk = tk & 1, t = tk >> 1;
    int hi = lane >> 4, n = lane & 15;
    int krow = 32 * kk + 8 * hi + e + (e >= 8 ? 8 : 0);
    Wsw[i] = f2bf(W[krow * K_ + 16 * t + n]);
  } else if (i < 40960) {                // 10*4 tiles * 32 lanes * 16 elems
    int j = i - 20480;
    int e = j & 15, lane = (j >> 4) & 31, tn = j >> 9;
    int nt = tn & 3, kt = tn >> 2;
    int hi = lane >> 4, n = lane & 15;
    int krow = 32 * kt + 8 * hi + e + (e >= 8 ? 8 : 0);
    Vsw[j] = f2bf(V[krow * D_ + 16 * nt + n]);
  }
}

// ---------------------------------------------------------------------------
// Kernel 2: fused VQ. blockDim = 64 (2 waves), one token per wave.
// ---------------------------------------------------------------------------
__launch_bounds__(32 * WPB)
__global__ void vq_main(const float* __restrict__ in,
                        const unsigned char* __restrict__ mask,
                        const float* __restrict__ bias,
                        const bf16* __restrict__ Wsw,
                        const bf16* __restrict__ Vsw,
                        float* __restrict__ cv_out,
                        int* __restrict__ cw_out) {
  __shared__ float lg[WPB][16][LP];     // per-wave logits [16 rows][320 cols]
  __shared__ float rmax[WPB][16];
  __shared__ float rsum[WPB][16];

  const int wave = threadIdx.x >> 5;
  const int lane = threadIdx.x & 31;
  const int tok  = blockIdx.x * WPB + wave;   // 0..16383
  const int hi   = lane >> 4;
  const int n    = lane & 15;

  // ---- A tile: 16 rows (= groups of this token) x 64, contiguous f32.
  // bf16 A layout: per 32-K step kk, elems 0..7 <- K = 32kk+8hi+c,
  //                                elems 8..15 <- K = 32kk+16+8hi+c.
  const float* rowp = in + ((size_t)tok * G_ + n) * (size_t)D_;
  float4 run[4][2];
#pragma unroll
  for (int j = 0; j < 4; ++j) {         // run j: K = 16j + 8hi + 0..7
    const float4* p = (const float4*)(rowp + 16 * j + 8 * hi);
    run[j][0] = p[0];
    run[j][1] = p[1];
  }
  v16bf A[2];
#pragma unroll
  for (int kk = 0; kk < 2; ++kk) {
    float av[16];
    float4 q0 = run[2 * kk][0],     q1 = run[2 * kk][1];
    float4 q2 = run[2 * kk + 1][0], q3 = run[2 * kk + 1][1];
    av[0]  = q0.x; av[1]  = q0.y; av[2]  = q0.z; av[3]  = q0.w;
    av[4]  = q1.x; av[5]  = q1.y; av[6]  = q1.z; av[7]  = q1.w;
    av[8]  = q2.x; av[9]  = q2.y; av[10] = q2.z; av[11] = q2.w;
    av[12] = q3.x; av[13] = q3.y; av[14] = q3.z; av[15] = q3.w;
#pragma unroll
    for (int e = 0; e < 16; ++e) A[kk][e] = f2bf(av[e]);
  }

  // ---- GEMM1: logits [16 x 320] -> LDS (f32), bias added.
#pragma unroll 1
  for (int t = 0; t < 20; ++t) {
    v16bf b0 = *(const v16bf*)(Wsw + ((size_t)(t * 2 + 0) * 32 + lane) * 16);
    v16bf b1 = *(const v16bf*)(Wsw + ((size_t)(t * 2 + 1) * 32 + lane) * 16);
    v8f acc = {};
    acc = __builtin_amdgcn_wmma_f32_16x16x32_bf16(false, A[0], false, b0,
                                                  (short)0, acc, false, false);
    acc = __builtin_amdgcn_wmma_f32_16x16x32_bf16(false, A[1], false, b1,
                                                  (short)0, acc, false, false);
    float bs = bias[t * 16 + n];
#pragma unroll
    for (int r = 0; r < 8; ++r)
      lg[wave][hi * 8 + r][t * 16 + n] = acc[r] + bs;   // D: row 8hi+r, col t*16+n
  }
  __syncthreads();

  // ---- softmax stats + argmax: 2 lanes per row, 160 cols each.
  const int row = lane >> 1, part = lane & 1;
  const float* lrow = &lg[wave][row][0];
  float m = -3.4e38f; int am = 0;
  for (int c = part * 160; c < part * 160 + 160; ++c) {
    float v = lrow[c];
    if (v > m) { m = v; am = c; }
  }
  float om = __shfl_xor(m, 1, 32);
  int   oa = __shfl_xor(am, 1, 32);
  if (om > m || (om == m && oa < am)) { m = om; am = oa; }   // first-max tie-break
  float s = 0.f;
  for (int c = part * 160; c < part * 160 + 160; ++c)
    s += __expf(lrow[c] - m);
  s += __shfl_xor(s, 1, 32);

  const bool mk = mask[tok] != 0;
  if (part == 0) {
    rmax[wave][row] = m;
    rsum[wave][row] = s;
    cw_out[(size_t)tok * G_ + row] = mk ? am : 0;       // codewords output
  }
  __syncthreads();

  // ---- GEMM2: cv = exp(logits - max) @ codevectors, scaled by 1/sum at end.
  const float mrow = rmax[wave][n];                     // this lane's A-row max
  v8f O[4] = {v8f{}, v8f{}, v8f{}, v8f{}};
#pragma unroll 1
  for (int kt = 0; kt < 10; ++kt) {
    const float* lr = &lg[wave][n][32 * kt + 8 * hi];
    float4 p0 = *(const float4*)(lr);        // cols +0..3
    float4 p1 = *(const float4*)(lr + 4);    // +4..7
    float4 p2 = *(const float4*)(lr + 16);   // +16..19
    float4 p3 = *(const float4*)(lr + 20);   // +20..23
    float pv[16];
    pv[0]  = p0.x; pv[1]  = p0.y; pv[2]  = p0.z; pv[3]  = p0.w;
    pv[4]  = p1.x; pv[5]  = p1.y; pv[6]  = p1.z; pv[7]  = p1.w;
    pv[8]  = p2.x; pv[9]  = p2.y; pv[10] = p2.z; pv[11] = p2.w;
    pv[12] = p3.x; pv[13] = p3.y; pv[14] = p3.z; pv[15] = p3.w;
    v16bf P;
#pragma unroll
    for (int e = 0; e < 16; ++e) P[e] = f2bf(__expf(pv[e] - mrow));
#pragma unroll
    for (int nt = 0; nt < 4; ++nt) {
      v16bf bv = *(const v16bf*)(Vsw + ((size_t)(kt * 4 + nt) * 32 + lane) * 16);
      O[nt] = __builtin_amdgcn_wmma_f32_16x16x32_bf16(false, P, false, bv,
                                                      (short)0, O[nt], false, false);
    }
  }

  // ---- epilogue: scale rows by 1/sum, apply mask, store cv.
#pragma unroll
  for (int r = 0; r < 8; ++r) {
    int rowm = hi * 8 + r;
    float sc = mk ? (1.0f / rsum[wave][rowm]) : 0.0f;
    size_t base = ((size_t)tok * G_ + rowm) * (size_t)D_ + n;
#pragma unroll
    for (int nt = 0; nt < 4; ++nt)
      cv_out[base + nt * 16] = O[nt][r] * sc;
  }
}

// ---------------------------------------------------------------------------
extern "C" void kernel_launch(void* const* d_in, const int* in_sizes, int n_in,
                              void* d_out, int out_size, void* d_ws, size_t ws_size,
                              hipStream_t stream) {
  (void)in_sizes; (void)n_in; (void)out_size; (void)ws_size;
  const float*         x    = (const float*)d_in[0];         // [B,S,H] f32
  const unsigned char* mask = (const unsigned char*)d_in[1]; // [B,S] bool
  const float*         W    = (const float*)d_in[2];         // [64,320] f32
  const float*         bias = (const float*)d_in[3];         // [320] f32
  const float*         V    = (const float*)d_in[4];         // [320,64] f32

  bf16* Wsw = (bf16*)d_ws;                                   // 40960 B
  bf16* Vsw = (bf16*)((char*)d_ws + 40960);                  // 40960 B

  float* cv = (float*)d_out;                                 // [B,S,H] f32
  int*   cw = (int*)((float*)d_out + (size_t)NROW * D_);     // [B,S,G] i32

  vq_swizzle<<<160, 256, 0, stream>>>(W, V, Wsw, Vsw);
  vq_main<<<NTOK / WPB, 32 * WPB, 0, stream>>>(x, mask, bias, Wsw, Vsw, cv, cw);
}
